// LocalGraphLearner_59923383713918
// MI455X (gfx1250) — matrix operand order repaired and checked by reading, not compiled, not run on hardware
//
#include <hip/hip_runtime.h>
#include <hip/hip_bf16.h>

#define B_   32
#define N_   1024
#define D_   256
#define KNN_ 16

typedef __bf16 bf16_t;
typedef __attribute__((ext_vector_type(16))) __bf16 v16bf;
typedef __attribute__((ext_vector_type(8)))  __bf16 v8bf;
typedef __attribute__((ext_vector_type(4)))  __bf16 v4bf;
typedef __attribute__((ext_vector_type(8)))  float  v8f;
typedef __attribute__((ext_vector_type(4)))  float  v4f;

// round-to-nearest-even f32 -> bf16 (bit-level, no __bf16 arithmetic needed)
__device__ __forceinline__ bf16_t to_bf16(float f) {
    unsigned u = __float_as_uint(f);
    unsigned r = (u + 0x7fffu + ((u >> 16) & 1u)) >> 16;
    unsigned short h = (unsigned short)r;
    return __builtin_bit_cast(bf16_t, h);
}

__device__ __forceinline__ float wave_max(float v) {
#pragma unroll
    for (int off = 16; off > 0; off >>= 1) v = fmaxf(v, __shfl_xor(v, off, 32));
    return v;
}
__device__ __forceinline__ float wave_sum(float v) {
#pragma unroll
    for (int off = 16; off > 0; off >>= 1) v += __shfl_xor(v, off, 32);
    return v;
}

// ---------------------------------------------------------------------------
// Kernel 1: h = x + pos_emb  ->  bf16   (B*N*D elements, 4 per thread)
// ---------------------------------------------------------------------------
__global__ void __launch_bounds__(256) k_prep_h(const float* __restrict__ x,
                                                const float* __restrict__ pos,
                                                bf16_t* __restrict__ hb) {
    size_t i = ((size_t)blockIdx.x * blockDim.x + threadIdx.x) * 4;
    v4f xv = *(const v4f*)(x + i);
    v4f pv = *(const v4f*)(pos + (i & (size_t)(N_ * D_ - 1)));
    v4bf o;
#pragma unroll
    for (int j = 0; j < 4; ++j) o[j] = to_bf16(xv[j] + pv[j]);
    *(v4bf*)(hb + i) = o;
}

// ---------------------------------------------------------------------------
// Kernel 2: convert both weight matrices to bf16 (layout [e,d] kept: this is
// exactly the N-major-by-K "B as B^T" layout WMMA's B operand wants)
// ---------------------------------------------------------------------------
__global__ void __launch_bounds__(256) k_prep_w(const float* __restrict__ wq,
                                                const float* __restrict__ wk,
                                                bf16_t* __restrict__ wqb,
                                                bf16_t* __restrict__ wkb) {
    size_t i = ((size_t)blockIdx.x * blockDim.x + threadIdx.x) * 4;
    v4f a = *(const v4f*)(wq + i);
    v4f b = *(const v4f*)(wk + i);
    v4bf oa, ob;
#pragma unroll
    for (int j = 0; j < 4; ++j) { oa[j] = to_bf16(a[j]); ob[j] = to_bf16(b[j]); }
    *(v4bf*)(wqb + i) = oa;
    *(v4bf*)(wkb + i) = ob;
}

// ---------------------------------------------------------------------------
// Kernel 3: projection  out = h @ w^T + bias   (bf16 in, bf16 out, f32 accum)
// grid.x = M/16 = 2048 blocks; block = 256 threads = 8 waves; wave w owns
// output columns [w*32, w*32+32) for this 16-row tile. K = 256 = 8 steps.
// ---------------------------------------------------------------------------
__global__ void __launch_bounds__(256) k_proj(const bf16_t* __restrict__ hb,
                                              const bf16_t* __restrict__ w,
                                              const float*  __restrict__ bias,
                                              bf16_t* __restrict__ outp) {
    const int m0   = blockIdx.x * 16;
    const int wave = threadIdx.x >> 5;
    const int lane = threadIdx.x & 31;
    const int r    = lane & 15;
    const int half = lane >> 4;
    const int n0   = wave * 32;

    // A operand (16x32 bf16): per lane two 8-elem chunks at K offsets h*16, h*16+8
    const bf16_t* pa = hb + (size_t)(m0 + r) * D_ + half * 16;
    // B operand (32x16 bf16, stored N-major x K): per lane one contiguous 16-elem
    // chunk at K offset h*16; column = lane&15
    const bf16_t* pb0 = w + (size_t)(n0 + r) * D_ + half * 16;
    const bf16_t* pb1 = pb0 + (size_t)16 * D_;

    v8f acc0 = {0,0,0,0,0,0,0,0};
    v8f acc1 = {0,0,0,0,0,0,0,0};

#pragma unroll
    for (int kk = 0; kk < D_; kk += 32) {
        v8bf lo = *(const v8bf*)(pa + kk);
        v8bf hi = *(const v8bf*)(pa + kk + 8);
        v16bf a;
#pragma unroll
        for (int i = 0; i < 8; ++i) { a[i] = lo[i]; a[8 + i] = hi[i]; }
        v16bf b0 = *(const v16bf*)(pb0 + kk);
        v16bf b1 = *(const v16bf*)(pb1 + kk);
        acc0 = __builtin_amdgcn_wmma_f32_16x16x32_bf16(false, a, false, b0,
                                                       (short)0, acc0, false, false);
        acc1 = __builtin_amdgcn_wmma_f32_16x16x32_bf16(false, a, false, b1,
                                                       (short)0, acc1, false, false);
    }

    // C/D layout: lane(0-15) -> N=lane, rows M = vgpr; lane(16-31) -> M = 8+vgpr
    const int col  = n0 + (lane & 15);
    const int row0 = m0 + half * 8;
    const float bi0 = bias[col];
    const float bi1 = bias[col + 16];
#pragma unroll
    for (int rr = 0; rr < 8; ++rr) {
        outp[(size_t)(row0 + rr) * D_ + col]      = to_bf16(acc0[rr] + bi0);
        outp[(size_t)(row0 + rr) * D_ + col + 16] = to_bf16(acc1[rr] + bi1);
    }
}

// ---------------------------------------------------------------------------
// Kernel 4: S = q @ k^T / sqrt(D) -> softmax -> top-16 sparsify -> dense out.
// grid = (N/16, B); block = 256 = 8 waves. Each block: 16 query rows x all
// 1024 key columns of one batch. Wave w owns key columns [w*128, w*128+128):
// 8 WMMA tiles x 8 K-steps = 64 WMMAs/wave. Scores staged in 64KB LDS, then
// fused softmax + top-k, full coalesced row writes (zeros + 16 values).
// ---------------------------------------------------------------------------
__global__ void __launch_bounds__(256) k_attn(const bf16_t* __restrict__ qb,
                                              const bf16_t* __restrict__ kb,
                                              float* __restrict__ out) {
    __shared__ float S[16][N_];   // 64 KB

    const int tile = blockIdx.x;      // query-row tile within batch (0..63)
    const int bidx = blockIdx.y;      // batch (0..31)
    const int wave = threadIdx.x >> 5;
    const int lane = threadIdx.x & 31;
    const int r    = lane & 15;
    const int half = lane >> 4;
    const int row0g = tile * 16;

    const size_t bbase = (size_t)bidx * N_;
    const bf16_t* pa = qb + (bbase + row0g + r) * D_ + half * 16;
    const bf16_t* pb = kb + (bbase + wave * 128 + r) * D_ + half * 16;

    v8f acc[8];
#pragma unroll
    for (int t = 0; t < 8; ++t) acc[t] = (v8f){0,0,0,0,0,0,0,0};

#pragma unroll
    for (int kk = 0; kk < D_; kk += 32) {
        v8bf lo = *(const v8bf*)(pa + kk);
        v8bf hi = *(const v8bf*)(pa + kk + 8);
        v16bf a;
#pragma unroll
        for (int i = 0; i < 8; ++i) { a[i] = lo[i]; a[8 + i] = hi[i]; }
#pragma unroll
        for (int t = 0; t < 8; ++t) {
            v16bf bf = *(const v16bf*)(pb + (size_t)t * 16 * D_ + kk);
            acc[t] = __builtin_amdgcn_wmma_f32_16x16x32_bf16(false, a, false, bf,
                                                             (short)0, acc[t], false, false);
        }
    }

    // stage scaled scores: S[row][col], scale = 1/sqrt(256) = 0.0625
#pragma unroll
    for (int t = 0; t < 8; ++t) {
        const int col = wave * 128 + t * 16 + (lane & 15);
#pragma unroll
        for (int rr = 0; rr < 8; ++rr)
            S[half * 8 + rr][col] = acc[t][rr] * 0.0625f;
    }
    __syncthreads();

    // softmax + top-16 per row; wave w handles rows 2w, 2w+1
    for (int rsel = 0; rsel < 2; ++rsel) {
        const int row = wave * 2 + rsel;
        float* rs = &S[row][0];

        // row max
        float m = -1e30f;
        for (int i = lane; i < N_; i += 32) m = fmaxf(m, rs[i]);
        m = wave_max(m);

        // exp in place + denominator over ALL entries
        float s = 0.f;
        for (int i = lane; i < N_; i += 32) {
            float e = __expf(rs[i] - m);
            rs[i] = e;
            s += e;
        }
        s = wave_sum(s);
        const float inv = 1.0f / s;

        // 16 argmax sweeps; winners marked negative in LDS (exp values are > 0)
        volatile float* vrs = rs;
        for (int t = 0; t < KNN_; ++t) {
            float bv = -1.f;
            int   bi = lane;
            for (int i = lane; i < N_; i += 32) {
                float v = vrs[i];
                if (v > bv) { bv = v; bi = i; }
            }
#pragma unroll
            for (int off = 16; off > 0; off >>= 1) {
                float ov = __shfl_xor(bv, off, 32);
                int   oi = __shfl_xor(bi, off, 32);
                if (ov > bv || (ov == bv && oi < bi)) { bv = ov; bi = oi; }
            }
            if (lane == 0) vrs[bi] = -fmaxf(bv, 1e-30f);
        }

        // full coalesced row write: zeros everywhere except the 16 winners
        float* orow = out + ((size_t)bidx * N_ + row0g + row) * (size_t)N_;
        for (int i = lane; i < N_; i += 32) {
            float v = vrs[i];
            orow[i] = (v < 0.f) ? (-v) * inv : 0.f;
        }
    }
}

// ---------------------------------------------------------------------------
extern "C" void kernel_launch(void* const* d_in, const int* in_sizes, int n_in,
                              void* d_out, int out_size, void* d_ws, size_t ws_size,
                              hipStream_t stream) {
    const float* x   = (const float*)d_in[0];
    const float* pos = (const float*)d_in[1];
    const float* wq  = (const float*)d_in[2];
    const float* bq  = (const float*)d_in[3];
    const float* wk  = (const float*)d_in[4];
    const float* bk  = (const float*)d_in[5];
    float* out = (float*)d_out;

    char* ws = (char*)d_ws;
    const size_t HSZ = (size_t)B_ * N_ * D_ * sizeof(bf16_t);   // 16 MB
    bf16_t* hb  = (bf16_t*)(ws);
    bf16_t* qb  = (bf16_t*)(ws + HSZ);
    bf16_t* kb  = (bf16_t*)(ws + 2 * HSZ);
    bf16_t* wqb = (bf16_t*)(ws + 3 * HSZ);
    bf16_t* wkb = (bf16_t*)(ws + 3 * HSZ + (size_t)D_ * D_ * sizeof(bf16_t));

    // 1) h = x + pos_emb -> bf16            (B*N*D / 4 per thread)
    k_prep_h<<<(B_ * N_ * D_) / (256 * 4), 256, 0, stream>>>(x, pos, hb);
    // 2) weights -> bf16
    k_prep_w<<<(D_ * D_) / (256 * 4), 256, 0, stream>>>(wq, wk, wqb, wkb);
    // 3) q = h @ wq^T + bq ; k = h @ wk^T + bk
    k_proj<<<(B_ * N_) / 16, 256, 0, stream>>>(hb, wqb, bq, qb);
    k_proj<<<(B_ * N_) / 16, 256, 0, stream>>>(hb, wkb, bk, kb);
    // 4) softmax(q k^T / sqrt(D)) -> top-16 sparse dense output
    k_attn<<<dim3(N_ / 16, B_), 256, 0, stream>>>(qb, kb, out);
}
